// CrossAttention_23278722744839
// MI455X (gfx1250) — matrix operand order repaired and checked by reading, compiled, not run on hardware
//
#include <hip/hip_runtime.h>
#include <hip/hip_bf16.h>

// ---------------------------------------------------------------------------
// CDNA5 (gfx1250) fused cross-attention, bf16 WMMA path + async LDS staging.
// Shapes fixed by the reference: b=2, dim=256, heads=8, d=64, seq=4096,
// context 32x32 -> bilinear 64x64.
// ---------------------------------------------------------------------------

typedef __attribute__((ext_vector_type(16))) __bf16 v16bf;
typedef __attribute__((ext_vector_type(8)))  float  v8f;
typedef int vsi4 __attribute__((vector_size(16)));   // matches builtin's int4

#define WMMA_BF16(A, B, C) \
  __builtin_amdgcn_wmma_f32_16x16x32_bf16(false, (A), false, (B), (short)0, (C), false, false)

#if defined(__gfx1250__) && __has_builtin(__builtin_amdgcn_global_load_async_to_lds_b128) && \
    __has_builtin(__builtin_amdgcn_s_wait_asynccnt)
#define HAVE_ASYNC_LDS 1
#else
#define HAVE_ASYNC_LDS 0
#endif

static __device__ __forceinline__ unsigned short f2bf(float f) {
  unsigned u = __float_as_uint(f);
  u = u + 0x7FFFu + ((u >> 16) & 1u);   // round-to-nearest-even
  return (unsigned short)(u >> 16);
}

// 16-byte global -> LDS copy (async on CDNA5, sync fallback otherwise)
static __device__ __forceinline__ void copy16_g2l(unsigned short* ldst,
                                                  const unsigned short* gsrc) {
#if HAVE_ASYNC_LDS
  __builtin_amdgcn_global_load_async_to_lds_b128(
      (__attribute__((address_space(1))) vsi4*)gsrc,
      (__attribute__((address_space(3))) vsi4*)ldst, 0, 0);
#else
  *(uint4*)ldst = *(const uint4*)gsrc;
#endif
}
static __device__ __forceinline__ void async_wait_pending2() {
#if HAVE_ASYNC_LDS
  __builtin_amdgcn_s_wait_asynccnt(2);
#endif
}
static __device__ __forceinline__ void async_wait_all() {
#if HAVE_ASYNC_LDS
  __builtin_amdgcn_s_wait_asynccnt(0);
#endif
}

union Frag16 {
  v16bf v;
  uint4 q[2];
  unsigned short u[16];
};

// A-fragment (16x32 bf16, M x K): row_ptr points at row M = lane&15.
// lanes<16: K = {kb+0..7, kb+16..23}; lanes>=16: K = {kb+8..15, kb+24..31}.
static __device__ __forceinline__ v16bf load_a_frag(const unsigned short* row_ptr,
                                                    int kbase, bool hi) {
  Frag16 f;
  const unsigned short* p0 = row_ptr + kbase + (hi ? 8 : 0);
  const unsigned short* p1 = row_ptr + kbase + 16 + (hi ? 8 : 0);
  f.q[0] = *(const uint4*)p0;
  f.q[1] = *(const uint4*)p1;
  return f.v;
}

// B-fragment (32x16 bf16, K x N) from column-major storage: row_ptr points at
// the K-run of column N = lane&15. lanes<16: K = {kb+0..7, kb+16..23}; lanes>=16: K = {kb+8..15, kb+24..31}.
static __device__ __forceinline__ v16bf load_b_frag(const unsigned short* row_ptr,
                                                    int kbase, bool hi) {
  Frag16 f;
  const unsigned short* p0 = row_ptr + kbase + (hi ? 8 : 0);
  const unsigned short* p1 = row_ptr + kbase + 16 + (hi ? 8 : 0);
  f.q[0] = *(const uint4*)p0;
  f.q[1] = *(const uint4*)p1;
  return f.v;
}

// ---------------------------------------------------------------------------
// f32 -> bf16 weight conversion
// ---------------------------------------------------------------------------
__global__ void f32_to_bf16_kernel(const float* __restrict__ s,
                                   unsigned short* __restrict__ d, int n) {
  int i = blockIdx.x * blockDim.x + threadIdx.x;
  if (i < n) d[i] = f2bf(s[i]);
}

// ---------------------------------------------------------------------------
// GroupNorm on x (2,256,64,64) -> bf16 pixel-major [b][p=4096][c=256]
// ---------------------------------------------------------------------------
__global__ void gn_x_kernel(const float* __restrict__ x, const float* __restrict__ w,
                            const float* __restrict__ bias, unsigned short* __restrict__ xnt) {
  const int C = 256, HW = 4096, CPG = 8, G = 32;
  int bb = blockIdx.x / G, g = blockIdx.x % G;
  const int N = CPG * HW;
  const float* xg = x + (size_t)(bb * C + g * CPG) * HW;
  float s = 0.f, sq = 0.f;
  for (int i = threadIdx.x; i < N; i += blockDim.x) {
    float v = xg[i]; s += v; sq += v * v;
  }
  __shared__ float ssum[256], ssq[256];
  ssum[threadIdx.x] = s; ssq[threadIdx.x] = sq;
  __syncthreads();
  for (int o = 128; o; o >>= 1) {
    if ((int)threadIdx.x < o) {
      ssum[threadIdx.x] += ssum[threadIdx.x + o];
      ssq[threadIdx.x]  += ssq[threadIdx.x + o];
    }
    __syncthreads();
  }
  float mean = ssum[0] / (float)N;
  float var  = ssq[0] / (float)N - mean * mean;
  float inv  = rsqrtf(var + 1e-5f);
  for (int i = threadIdx.x; i < N; i += blockDim.x) {
    int cl = i / HW, p = i % HW;
    int c = g * CPG + cl;
    float v = (xg[i] - mean) * inv * w[c] + bias[c];
    xnt[(size_t)(bb * HW + p) * C + c] = f2bf(v);
  }
}

// ---------------------------------------------------------------------------
// GroupNorm on context (2,256,32,32) -> f32 channel-major [b][c][1024]
// ---------------------------------------------------------------------------
__global__ void gn_ctx_kernel(const float* __restrict__ x, const float* __restrict__ w,
                              const float* __restrict__ bias, float* __restrict__ outc) {
  const int C = 256, HW = 1024, CPG = 8, G = 32;
  int bb = blockIdx.x / G, g = blockIdx.x % G;
  const int N = CPG * HW;
  const float* xg = x + (size_t)(bb * C + g * CPG) * HW;
  float s = 0.f, sq = 0.f;
  for (int i = threadIdx.x; i < N; i += blockDim.x) {
    float v = xg[i]; s += v; sq += v * v;
  }
  __shared__ float ssum[256], ssq[256];
  ssum[threadIdx.x] = s; ssq[threadIdx.x] = sq;
  __syncthreads();
  for (int o = 128; o; o >>= 1) {
    if ((int)threadIdx.x < o) {
      ssum[threadIdx.x] += ssum[threadIdx.x + o];
      ssq[threadIdx.x]  += ssq[threadIdx.x + o];
    }
    __syncthreads();
  }
  float mean = ssum[0] / (float)N;
  float var  = ssq[0] / (float)N - mean * mean;
  float inv  = rsqrtf(var + 1e-5f);
  for (int i = threadIdx.x; i < N; i += blockDim.x) {
    int cl = i / HW, p = i % HW;
    int c = g * CPG + cl;
    outc[(size_t)(bb * C + c) * HW + p] = (xg[i] - mean) * inv * w[c] + bias[c];
  }
}

// ---------------------------------------------------------------------------
// Bilinear resize 32x32 -> 64x64 (half-pixel centers, clamped), write bf16
// pixel-major [b][p=4096][c=256]
// ---------------------------------------------------------------------------
__global__ void resize_kernel(const float* __restrict__ ctxn,
                              unsigned short* __restrict__ ctxt) {
  int idx = blockIdx.x * blockDim.x + threadIdx.x;
  const int total = 2 * 4096 * 256;
  if (idx >= total) return;
  int c  = idx % 256;
  int p  = (idx / 256) % 4096;
  int bb = idx / (256 * 4096);
  int y = p >> 6, x = p & 63;
  float sy = y * 0.5f - 0.25f, sx = x * 0.5f - 0.25f;
  int y0 = (int)floorf(sy), x0 = (int)floorf(sx);
  float fy = sy - (float)y0, fx = sx - (float)x0;
  int y0c = min(max(y0, 0), 31), y1c = min(max(y0 + 1, 0), 31);
  int x0c = min(max(x0, 0), 31), x1c = min(max(x0 + 1, 0), 31);
  const float* src = ctxn + (size_t)(bb * 256 + c) * 1024;
  float v00 = src[y0c * 32 + x0c], v01 = src[y0c * 32 + x1c];
  float v10 = src[y1c * 32 + x0c], v11 = src[y1c * 32 + x1c];
  float v = (1.f - fy) * ((1.f - fx) * v00 + fx * v01) +
            fy * ((1.f - fx) * v10 + fx * v11);
  ctxt[idx] = f2bf(v);
}

// ---------------------------------------------------------------------------
// Q projection: Wq(512x256) x xn_t(b,4096,256)^T -> q[b][h][seq][64] bf16
// ---------------------------------------------------------------------------
__global__ __launch_bounds__(256) void proj_q_kernel(const unsigned short* __restrict__ Wbf,
                                                     const unsigned short* __restrict__ Bt,
                                                     unsigned short* __restrict__ qout) {
  int wid = threadIdx.x >> 5, lane = threadIdx.x & 31;
  bool hi = lane >= 16; int l15 = lane & 15;
  int tile = blockIdx.x * 8 + wid;          // 2 * 32 * 256 tiles
  int nt = tile % 256, mt = (tile / 256) % 32, bb = tile / (256 * 32);
  const unsigned short* arow = Wbf + (size_t)(mt * 16 + l15) * 256;
  const unsigned short* brow = Bt + (size_t)(bb * 4096 + nt * 16 + l15) * 256;
  v8f acc = {};
#pragma unroll
  for (int k = 0; k < 256; k += 32) {
    v16bf a = load_a_frag(arow, k, hi);
    v16bf b = load_b_frag(brow, k, hi);
    acc = WMMA_BF16(a, b, acc);
  }
  int p = nt * 16 + l15;
  int o0 = mt * 16 + (hi ? 8 : 0);
  int head = o0 >> 6, d0 = o0 & 63;
  union { uint4 q; unsigned short u[8]; } st;
#pragma unroll
  for (int r = 0; r < 8; r++) st.u[r] = f2bf(acc[r]);
  *(uint4*)(qout + ((size_t)(bb * 8 + head) * 4096 + p) * 64 + d0) = st.q;
}

// ---------------------------------------------------------------------------
// KV projection: Wkv(1024x256) x ctx_t^T. First 512 rows -> k[b][h][seq][64],
// last 512 -> vt[b][h][64][seq] (transposed for PV B-fragments).
// ---------------------------------------------------------------------------
__global__ __launch_bounds__(256) void proj_kv_kernel(const unsigned short* __restrict__ Wbf,
                                                      const unsigned short* __restrict__ Bt,
                                                      unsigned short* __restrict__ kout,
                                                      unsigned short* __restrict__ vtout) {
  int wid = threadIdx.x >> 5, lane = threadIdx.x & 31;
  bool hi = lane >= 16; int l15 = lane & 15;
  int tile = blockIdx.x * 8 + wid;          // 2 * 64 * 256 tiles
  int nt = tile % 256, mt = (tile / 256) % 64, bb = tile / (256 * 64);
  const unsigned short* arow = Wbf + (size_t)(mt * 16 + l15) * 256;
  const unsigned short* brow = Bt + (size_t)(bb * 4096 + nt * 16 + l15) * 256;
  v8f acc = {};
#pragma unroll
  for (int k = 0; k < 256; k += 32) {
    v16bf a = load_a_frag(arow, k, hi);
    v16bf b = load_b_frag(brow, k, hi);
    acc = WMMA_BF16(a, b, acc);
  }
  int p = nt * 16 + l15;
  int o0 = mt * 16 + (hi ? 8 : 0);
  if (o0 < 512) {                            // K path, row-major [h][seq][64]
    int head = o0 >> 6, d0 = o0 & 63;
    union { uint4 q; unsigned short u[8]; } st;
#pragma unroll
    for (int r = 0; r < 8; r++) st.u[r] = f2bf(acc[r]);
    *(uint4*)(kout + ((size_t)(bb * 8 + head) * 4096 + p) * 64 + d0) = st.q;
  } else {                                   // V path, transposed [h][64][seq]
    int ov = o0 - 512;
    int head = ov >> 6, d0 = ov & 63;
#pragma unroll
    for (int r = 0; r < 8; r++)
      vtout[((size_t)(bb * 8 + head) * 64 + d0 + r) * 4096 + p] = f2bf(acc[r]);
  }
}

// ---------------------------------------------------------------------------
// Flash attention with double-buffered async K/V staging in LDS.
// Block = 8 waves = 128 q rows for one (b,h). Per 32-key step the block stages
// K tile (32x64) and V^T tile (64x32) once via GLOBAL_LOAD_ASYNC_TO_LDS_B128;
// each wave then does 4 S-wmma + online softmax + 4 PV-wmma from LDS.
// Rows padded (72/40 ushorts) for 16B-aligned, bank-rotated ds_load_b128.
// ---------------------------------------------------------------------------
__global__ __launch_bounds__(256) void attn_kernel(const unsigned short* __restrict__ q,
                                                   const unsigned short* __restrict__ k,
                                                   const unsigned short* __restrict__ vt,
                                                   unsigned short* __restrict__ obuf) {
  const int SEQ = 4096, D = 64, H = 8;
  int tid = threadIdx.x;
  int wid = tid >> 5, lane = tid & 31;
  bool hi = lane >= 16; int l15 = lane & 15;
  int qb = blockIdx.x % 32;
  int h  = (blockIdx.x / 32) % H;
  int bb = blockIdx.x / (32 * H);

  const unsigned short* Qh = q + (size_t)(bb * H + h) * SEQ * D;
  const unsigned short* Kh = k + (size_t)(bb * H + h) * SEQ * D;
  const unsigned short* Vh = vt + (size_t)(bb * H + h) * D * SEQ;

  __shared__ __align__(16) unsigned short Ks[2][32][72];     // K rows, padded
  __shared__ __align__(16) unsigned short Vs[2][64][40];     // V^T rows, padded
  __shared__ __align__(16) unsigned short pshare[8][16][40]; // per-wave P slab

  // staging assignment: 256 threads x 16B cover each 4KB tile
  int krow = tid >> 3, kcol = (tid & 7) * 8;   // K: 32 rows x 64 cols
  int vrow = tid >> 2, vcol = (tid & 3) * 8;   // V: 64 rows x 32 cols
  auto stage = [&](int j0, int bufi) {
    copy16_g2l(&Ks[bufi][krow][kcol], Kh + (size_t)(j0 + krow) * D + kcol);
    copy16_g2l(&Vs[bufi][vrow][vcol], Vh + (size_t)vrow * SEQ + j0 + vcol);
  };

  int qr0 = qb * 128 + wid * 16;
  const unsigned short* qrow = Qh + (size_t)(qr0 + l15) * D;
  v16bf aq0 = load_a_frag(qrow, 0, hi);
  v16bf aq1 = load_a_frag(qrow, 32, hi);

  v8f O0 = {}, O1 = {}, O2 = {}, O3 = {};
  float m[8], l[8];
#pragma unroll
  for (int r = 0; r < 8; r++) { m[r] = -3.0e38f; l[r] = 0.f; }

  unsigned short(*ps)[40] = pshare[wid];
  const float scale = 0.125f;   // 64^-0.5

  stage(0, 0);                                 // prologue: stage 0
  for (int it = 0; it < SEQ / 32; ++it) {
    int j0 = it * 32;
    int buf = it & 1;
    __syncthreads();                           // next buffer fully consumed
    bool has_next = (it + 1) < SEQ / 32;
    if (has_next) {
      stage(j0 + 32, (it + 1) & 1);
      async_wait_pending2();                   // stage `it` landed (in-order)
    } else {
      async_wait_all();
    }
    __syncthreads();                           // stage `it` visible to block

    // S = Q K^T for 32 keys (two 16-col tiles)
    v16bf b00 = load_b_frag(&Ks[buf][l15][0],      0,  hi);
    v16bf b01 = load_b_frag(&Ks[buf][l15][0],      32, hi);
    v16bf b10 = load_b_frag(&Ks[buf][16 + l15][0], 0,  hi);
    v16bf b11 = load_b_frag(&Ks[buf][16 + l15][0], 32, hi);
    v8f s0 = {}, s1 = {};
    s0 = WMMA_BF16(aq0, b00, s0);
    s0 = WMMA_BF16(aq1, b01, s0);
    s1 = WMMA_BF16(aq0, b10, s1);
    s1 = WMMA_BF16(aq1, b11, s1);

#pragma unroll
    for (int r = 0; r < 8; r++) {
      float a0 = s0[r] * scale, a1 = s1[r] * scale;
      float t = fmaxf(a0, a1);
      t = fmaxf(t, __shfl_xor(t, 8, 16));
      t = fmaxf(t, __shfl_xor(t, 4, 16));
      t = fmaxf(t, __shfl_xor(t, 2, 16));
      t = fmaxf(t, __shfl_xor(t, 1, 16));
      float mn = fmaxf(m[r], t);
      float alpha = __expf(m[r] - mn);
      m[r] = mn;
      float p0 = __expf(a0 - mn), p1 = __expf(a1 - mn);
      float rs = p0 + p1;
      rs += __shfl_xor(rs, 8, 16);
      rs += __shfl_xor(rs, 4, 16);
      rs += __shfl_xor(rs, 2, 16);
      rs += __shfl_xor(rs, 1, 16);
      l[r] = l[r] * alpha + rs;
      O0[r] *= alpha; O1[r] *= alpha; O2[r] *= alpha; O3[r] *= alpha;
      int Mrow = r + (hi ? 8 : 0);
      ps[Mrow][l15]      = f2bf(p0);
      ps[Mrow][16 + l15] = f2bf(p1);
    }

    // reload P in A-fragment layout (wave-local LDS transpose; DS in-order)
    v16bf pf = load_a_frag(&ps[l15][0], 0, hi);

    v16bf bv0 = load_b_frag(&Vs[buf][0 * 16 + l15][0], 0, hi);
    v16bf bv1 = load_b_frag(&Vs[buf][1 * 16 + l15][0], 0, hi);
    v16bf bv2 = load_b_frag(&Vs[buf][2 * 16 + l15][0], 0, hi);
    v16bf bv3 = load_b_frag(&Vs[buf][3 * 16 + l15][0], 0, hi);
    O0 = WMMA_BF16(pf, bv0, O0);
    O1 = WMMA_BF16(pf, bv1, O1);
    O2 = WMMA_BF16(pf, bv2, O2);
    O3 = WMMA_BF16(pf, bv3, O3);
  }

  // epilogue: normalize and store bf16 pixel-major [b][p][inner=512]
#pragma unroll
  for (int r = 0; r < 8; r++) {
    float inv = 1.0f / l[r];
    int row = qr0 + r + (hi ? 8 : 0);
    unsigned short* orow = obuf + (size_t)(bb * SEQ + row) * 512 + h * 64;
    orow[0 * 16 + l15] = f2bf(O0[r] * inv);
    orow[1 * 16 + l15] = f2bf(O1[r] * inv);
    orow[2 * 16 + l15] = f2bf(O2[r] * inv);
    orow[3 * 16 + l15] = f2bf(O3[r] * inv);
  }
}

// ---------------------------------------------------------------------------
// Output projection: Wout(256x512) x o(b,4096,512)^T + bout -> f32 (b,256,64,64)
// ---------------------------------------------------------------------------
__global__ __launch_bounds__(256) void proj_out_kernel(const unsigned short* __restrict__ Wbf,
                                                       const unsigned short* __restrict__ Bt,
                                                       const float* __restrict__ bout,
                                                       float* __restrict__ out) {
  int wid = threadIdx.x >> 5, lane = threadIdx.x & 31;
  bool hi = lane >= 16; int l15 = lane & 15;
  int tile = blockIdx.x * 8 + wid;          // 2 * 16 * 256 tiles
  int nt = tile % 256, mt = (tile / 256) % 16, bb = tile / (256 * 16);
  const unsigned short* arow = Wbf + (size_t)(mt * 16 + l15) * 512;
  const unsigned short* brow = Bt + (size_t)(bb * 4096 + nt * 16 + l15) * 512;
  v8f acc = {};
#pragma unroll
  for (int k = 0; k < 512; k += 32) {
    v16bf a = load_a_frag(arow, k, hi);
    v16bf b = load_b_frag(brow, k, hi);
    acc = WMMA_BF16(a, b, acc);
  }
  int p = nt * 16 + l15;
#pragma unroll
  for (int r = 0; r < 8; r++) {
    int o = mt * 16 + (hi ? 8 : 0) + r;
    out[(size_t)(bb * 256 + o) * 4096 + p] = acc[r] + bout[o];
  }
}

// ---------------------------------------------------------------------------
extern "C" void kernel_launch(void* const* d_in, const int* in_sizes, int n_in,
                              void* d_out, int out_size, void* d_ws, size_t ws_size,
                              hipStream_t stream) {
  const float* x       = (const float*)d_in[0];
  const float* context = (const float*)d_in[1];
  const float* gn_w    = (const float*)d_in[2];
  const float* gn_b    = (const float*)d_in[3];
  const float* gnc_w   = (const float*)d_in[4];
  const float* gnc_b   = (const float*)d_in[5];
  const float* Wq      = (const float*)d_in[6];
  const float* Wkv     = (const float*)d_in[7];
  const float* Wout    = (const float*)d_in[8];
  const float* bout    = (const float*)d_in[9];
  float* out = (float*)d_out;

  char* ws = (char*)d_ws;
  size_t off = 0;
  auto alloc = [&](size_t bytes) -> void* {
    void* p = ws + off;
    off = (off + bytes + 255) & ~(size_t)255;
    return p;
  };
  unsigned short* wq_bf   = (unsigned short*)alloc((size_t)512 * 256 * 2);
  unsigned short* wkv_bf  = (unsigned short*)alloc((size_t)1024 * 256 * 2);
  unsigned short* wout_bf = (unsigned short*)alloc((size_t)256 * 512 * 2);
  unsigned short* xnt     = (unsigned short*)alloc((size_t)2 * 4096 * 256 * 2);
  float*          ctxn    = (float*)         alloc((size_t)2 * 256 * 1024 * 4);
  unsigned short* ctxt    = (unsigned short*)alloc((size_t)2 * 4096 * 256 * 2);
  unsigned short* qb      = (unsigned short*)alloc((size_t)2 * 8 * 4096 * 64 * 2);
  unsigned short* kb      = (unsigned short*)alloc((size_t)2 * 8 * 4096 * 64 * 2);
  unsigned short* vtb     = (unsigned short*)alloc((size_t)2 * 8 * 64 * 4096 * 2);
  unsigned short* ob      = (unsigned short*)alloc((size_t)2 * 4096 * 512 * 2);
  (void)ws_size; (void)in_sizes; (void)n_in; (void)out_size;

  // 1) weight conversion to bf16
  f32_to_bf16_kernel<<<(512 * 256 + 255) / 256, 256, 0, stream>>>(Wq, wq_bf, 512 * 256);
  f32_to_bf16_kernel<<<(1024 * 256 + 255) / 256, 256, 0, stream>>>(Wkv, wkv_bf, 1024 * 256);
  f32_to_bf16_kernel<<<(256 * 512 + 255) / 256, 256, 0, stream>>>(Wout, wout_bf, 256 * 512);

  // 2) group norms
  gn_x_kernel<<<64, 256, 0, stream>>>(x, gn_w, gn_b, xnt);
  gn_ctx_kernel<<<64, 256, 0, stream>>>(context, gnc_w, gnc_b, ctxn);

  // 3) bilinear resize 32x32 -> 64x64, pixel-major bf16
  resize_kernel<<<(2 * 4096 * 256 + 255) / 256, 256, 0, stream>>>(ctxn, ctxt);

  // 4) projections (WMMA GEMMs)
  proj_q_kernel<<<(2 * 32 * 256) / 8, 256, 0, stream>>>(wq_bf, xnt, qb);
  proj_kv_kernel<<<(2 * 64 * 256) / 8, 256, 0, stream>>>(wkv_bf, ctxt, kb, vtb);

  // 5) flash attention
  attn_kernel<<<2 * 8 * 32, 256, 0, stream>>>(qb, kb, vtb, ob);

  // 6) output projection + bias
  proj_out_kernel<<<(2 * 16 * 256) / 8, 256, 0, stream>>>(wout_bf, ob, bout, out);
}
